// LRULayer_5952824673190
// MI455X (gfx1250) — matrix operand order repaired and checked by reading, not compiled, not run on hardware
//
#include <hip/hip_runtime.h>

// ---------------------------------------------------------------------------
// LRU layer (S5-style): y = Re(C @ scan(Lambda, B_norm @ x)) + D*x
// L=131072, N=256, H=256.  Chunked parallel scan, bf16 WMMA GEMMs.
// ---------------------------------------------------------------------------

#define LDIM 131072
#define HDIM 256
#define NDIM 256
#define CHUNK 256          // rows per block
#define NCHUNK 512         // LDIM / CHUNK
#define SUB 16             // rows per subtile
#define NSUB 16            // CHUNK / SUB

typedef __attribute__((ext_vector_type(16))) __bf16 v16bf;
typedef __attribute__((ext_vector_type(8)))  __bf16 v8bf;
typedef __attribute__((ext_vector_type(8)))  float  v8f;

__device__ inline v8f wmma_bf16(v16bf a, v16bf b, v8f c) {
    // D = A(16x32 bf16) * B(32x16 bf16) + C(16x16 f32)
    return __builtin_amdgcn_wmma_f32_16x16x32_bf16(
        /*neg_a=*/false, a, /*neg_b=*/false, b,
        /*c_mod=*/(short)0, c, /*reuse_a=*/false, /*reuse_b=*/false);
}

// A-fragment (16x32 bf16) from LDS, row-major [16][256] bf16.
// Lane l: row = l&15.  ISA layout: lanes<16 hold K={kb..kb+7, kb+16..kb+23},
// lanes>=16 hold K={kb+8..kb+15, kb+24..kb+31}  -> two contiguous b128 loads.
__device__ inline v16bf load_a_frag(const __bf16* lds, int lane, int kb) {
    int row = lane & 15;
    int hi  = (lane >> 4) << 3;              // 0 or 8
    const __bf16* p = lds + row * 256 + kb + hi;
    union { v16bf v; v8bf h[2]; } u;
    u.h[0] = *(const v8bf*)(p);              // elements 0..7  (K = kb+hi .. +7)
    u.h[1] = *(const v8bf*)(p + 16);         // elements 8..15 (K = kb+16+hi ..)
    return u.v;
}

// B-fragment from pre-swizzled global tile array:
// index = ((kt*16 + nt)*32 + lane)*16 + e  -> one contiguous 32B load per lane.
__device__ inline v16bf load_b_frag(const __bf16* tiled, int kt, int nt, int lane) {
    return *(const v16bf*)(tiled + ((((kt * 16 + nt) * 32) + lane) << 4));
}

// ---------------------------------------------------------------------------
// K0a: Lambda = exp(-exp(nu) + i exp(theta)); LambdaPow = Lambda^CHUNK
// ---------------------------------------------------------------------------
__global__ void k_prep_lambda(const float* __restrict__ nu_log,
                              const float* __restrict__ theta_log,
                              float* lam_re, float* lam_im,
                              float* lamT_re, float* lamT_im) {
    int n = threadIdx.x;
    float en = __expf(nu_log[n]);
    float th = __expf(theta_log[n]);
    float r  = __expf(-en);
    float lr = r * __cosf(th);
    float li = r * __sinf(th);
    lam_re[n] = lr;
    lam_im[n] = li;
    float ar = lr, ai = li;
#pragma unroll
    for (int i = 0; i < 8; ++i) {            // ^2 eight times -> ^256
        float nr = ar * ar - ai * ai;
        float ni = 2.f * ar * ai;
        ar = nr; ai = ni;
    }
    lamT_re[n] = ar;
    lamT_im[n] = ai;
}

// ---------------------------------------------------------------------------
// K0b: swizzle weights into WMMA-B fragment layout (bf16).
//   W[k=h][n]  = (B_re + i B_im)[n][h] * exp(gamma_log[n])   (input GEMM)
//   CT[k=n][h] =  C_re/C_im[h][n]                            (output GEMM)
// ---------------------------------------------------------------------------
__global__ void k_prep_tiles(const float* __restrict__ B_re,
                             const float* __restrict__ B_im,
                             const float* __restrict__ gamma_log,
                             const float* __restrict__ C_re,
                             const float* __restrict__ C_im,
                             __bf16* Wre, __bf16* Wim,
                             __bf16* CTre, __bf16* CTim) {
    int tid  = blockIdx.x * blockDim.x + threadIdx.x;   // 0 .. 65535
    int e    = tid & 15;
    int lane = (tid >> 4) & 31;
    int nt   = (tid >> 9) & 15;
    int kt   = tid >> 13;                                // 0..7
    int k = kt * 32 + ((lane >> 4) << 3) + (e & 7) + ((e >> 3) << 4);  // K of fragment slot
    int c = nt * 16 + (lane & 15);                       // column of fragment

    float g = __expf(gamma_log[c]);
    Wre[tid] = (__bf16)(B_re[c * HDIM + k] * g);         // B_re[n=c][h=k]
    Wim[tid] = (__bf16)(B_im[c * HDIM + k] * g);
    CTre[tid] = (__bf16)(C_re[c * NDIM + k]);            // C_re[h=c][n=k]
    CTim[tid] = (__bf16)(C_im[c * NDIM + k]);
}

// ---------------------------------------------------------------------------
// Shared subtile routine: stage 16x256 of x into LDS (bf16 + optional f32),
// then 8 waves compute Bu_re / Bu_im (16x256 each) with WMMA into LDS.
// Wave w owns tiles tt = 4w..4w+3: tt<16 -> Bu_re col-tile tt, else Bu_im.
// ---------------------------------------------------------------------------
template <bool WITH_XS>
__device__ inline void compute_bu_subtile(const float* __restrict__ xg,
                                          long rowbase,
                                          const __bf16* __restrict__ Wre,
                                          const __bf16* __restrict__ Wim,
                                          __bf16* xb, float* xs,
                                          float* bu_re, float* bu_im,
                                          int tid) {
    const float4* src = (const float4*)(xg + rowbase * HDIM);
    for (int i = tid; i < (SUB * HDIM) / 4; i += 256) {
        float4 v = src[i];
        int b0 = i * 4;
        xb[b0 + 0] = (__bf16)v.x; xb[b0 + 1] = (__bf16)v.y;
        xb[b0 + 2] = (__bf16)v.z; xb[b0 + 3] = (__bf16)v.w;
        if (WITH_XS) {
            xs[b0 + 0] = v.x; xs[b0 + 1] = v.y;
            xs[b0 + 2] = v.z; xs[b0 + 3] = v.w;
        }
    }
    __syncthreads();

    int lane = tid & 31, wave = tid >> 5;
    v8f acc[4] = {};
    for (int kt = 0; kt < 8; ++kt) {
        v16bf a = load_a_frag(xb, lane, kt * 32);
#pragma unroll
        for (int j = 0; j < 4; ++j) {
            int tt = wave * 4 + j;
            const __bf16* Wt = (tt & 16) ? Wim : Wre;
            v16bf b = load_b_frag(Wt, kt, tt & 15, lane);
            acc[j] = wmma_bf16(a, b, acc[j]);
        }
    }
    int rofs = (lane >> 4) << 3;             // C/D layout: M = v + 8*(lane>=16)
#pragma unroll
    for (int j = 0; j < 4; ++j) {
        int tt = wave * 4 + j;
        float* dst = (tt & 16) ? bu_im : bu_re;
        int col = (tt & 15) * 16 + (lane & 15);
#pragma unroll
        for (int v = 0; v < 8; ++v)
            dst[(v + rofs) * 256 + col] = acc[j][v];
    }
    __syncthreads();
}

// ---------------------------------------------------------------------------
// K1: per-chunk local scan (zero init) -> chunk carry b[g][n]
// ---------------------------------------------------------------------------
__global__ void __launch_bounds__(256)
k_chunk_carry(const float* __restrict__ x,
              const __bf16* __restrict__ Wre, const __bf16* __restrict__ Wim,
              const float* __restrict__ lam_re, const float* __restrict__ lam_im,
              float* __restrict__ b_re, float* __restrict__ b_im) {
    __shared__ __align__(16) __bf16 xb[SUB * 256];
    __shared__ __align__(16) float  bu_re[SUB * 256];
    __shared__ __align__(16) float  bu_im[SUB * 256];
    int tid = threadIdx.x;
    int g   = blockIdx.x;
    float lr = lam_re[tid], li = lam_im[tid];
    float zr = 0.f, zi = 0.f;

    for (int s = 0; s < NSUB; ++s) {
        long rowbase = (long)g * CHUNK + s * SUB;
        compute_bu_subtile<false>(x, rowbase, Wre, Wim, xb, nullptr,
                                  bu_re, bu_im, tid);
#pragma unroll
        for (int r = 0; r < SUB; ++r) {      // z = Lambda*z + Bu  (channel tid)
            float br = bu_re[r * 256 + tid], bi = bu_im[r * 256 + tid];
            float nr = lr * zr - li * zi + br;
            float ni = lr * zi + li * zr + bi;
            zr = nr; zi = ni;
        }
        __syncthreads();
    }
    b_re[g * NDIM + tid] = zr;
    b_im[g * NDIM + tid] = zi;
}

// ---------------------------------------------------------------------------
// K2: sequential prefix over 512 chunk carries (1 block, 1 channel/thread)
// ---------------------------------------------------------------------------
__global__ void k_prefix(const float* __restrict__ lamT_re,
                         const float* __restrict__ lamT_im,
                         const float* __restrict__ b_re,
                         const float* __restrict__ b_im,
                         float* __restrict__ p_re, float* __restrict__ p_im) {
    int t = threadIdx.x;
    float ar = lamT_re[t], ai = lamT_im[t];
    float zr = 0.f, zi = 0.f;
    for (int g = 0; g < NCHUNK; ++g) {
        p_re[g * NDIM + t] = zr;
        p_im[g * NDIM + t] = zi;
        float br = b_re[g * NDIM + t], bi = b_im[g * NDIM + t];
        float nr = ar * zr - ai * zi + br;
        float ni = ar * zi + ai * zr + bi;
        zr = nr; zi = ni;
    }
}

// ---------------------------------------------------------------------------
// K3: replay GEMM1 + scan seeded with prefix, output GEMM, y = Re(Cz) + D*x
// ---------------------------------------------------------------------------
__global__ void __launch_bounds__(256)
k_output(const float* __restrict__ x,
         const __bf16* __restrict__ Wre, const __bf16* __restrict__ Wim,
         const __bf16* __restrict__ CTre, const __bf16* __restrict__ CTim,
         const float* __restrict__ lam_re, const float* __restrict__ lam_im,
         const float* __restrict__ p_re, const float* __restrict__ p_im,
         const float* __restrict__ D, float* __restrict__ y) {
    __shared__ __align__(16) __bf16 xb[SUB * 256];
    __shared__ __align__(16) float  xs[SUB * 256];
    __shared__ __align__(16) float  bu_re[SUB * 256];
    __shared__ __align__(16) float  bu_im[SUB * 256];
    __shared__ __align__(16) __bf16 zbre[SUB * 256];
    __shared__ __align__(16) __bf16 zbim[SUB * 256];   // holds -Im(z) in bf16

    int tid  = threadIdx.x;
    int g    = blockIdx.x;
    int lane = tid & 31, wave = tid >> 5;
    float lr = lam_re[tid], li = lam_im[tid];
    float zr = p_re[g * NDIM + tid];
    float zi = p_im[g * NDIM + tid];

    for (int s = 0; s < NSUB; ++s) {
        long rowbase = (long)g * CHUNK + s * SUB;
        compute_bu_subtile<true>(x, rowbase, Wre, Wim, xb, xs,
                                 bu_re, bu_im, tid);
#pragma unroll
        for (int r = 0; r < SUB; ++r) {
            float br = bu_re[r * 256 + tid], bi = bu_im[r * 256 + tid];
            float nr = lr * zr - li * zi + br;
            float ni = lr * zi + li * zr + bi;
            zr = nr; zi = ni;
            zbre[r * 256 + tid] = (__bf16)zr;
            zbim[r * 256 + tid] = (__bf16)(-zi);       // fold the minus sign
        }
        __syncthreads();

        // y_sub = zre*CTre + (-zim)*CTim : wave owns H col-tiles {2w, 2w+1}
        v8f acc[2] = {};
        for (int kt = 0; kt < 8; ++kt) {
            v16bf are = load_a_frag(zbre, lane, kt * 32);
            v16bf aim = load_a_frag(zbim, lane, kt * 32);
#pragma unroll
            for (int j = 0; j < 2; ++j) {
                int ht = wave * 2 + j;
                acc[j] = wmma_bf16(are, load_b_frag(CTre, kt, ht, lane), acc[j]);
                acc[j] = wmma_bf16(aim, load_b_frag(CTim, kt, ht, lane), acc[j]);
            }
        }
        int rofs = (lane >> 4) << 3;
#pragma unroll
        for (int j = 0; j < 2; ++j) {
            int ht = wave * 2 + j;
            int h  = ht * 16 + (lane & 15);
            float dh = D[h];
#pragma unroll
            for (int v = 0; v < 8; ++v) {
                int r = v + rofs;
                y[(size_t)(rowbase + r) * HDIM + h] = acc[j][v] + dh * xs[r * 256 + h];
            }
        }
        __syncthreads();
    }
}

// ---------------------------------------------------------------------------
// Launch
// ---------------------------------------------------------------------------
extern "C" void kernel_launch(void* const* d_in, const int* in_sizes, int n_in,
                              void* d_out, int out_size, void* d_ws, size_t ws_size,
                              hipStream_t stream) {
    const float* x         = (const float*)d_in[0];
    const float* nu_log    = (const float*)d_in[1];
    const float* theta_log = (const float*)d_in[2];
    const float* B_re      = (const float*)d_in[3];
    const float* B_im      = (const float*)d_in[4];
    const float* C_re      = (const float*)d_in[5];
    const float* C_im      = (const float*)d_in[6];
    const float* Dv        = (const float*)d_in[7];
    const float* gamma_log = (const float*)d_in[8];
    float* y = (float*)d_out;

    // Workspace carve-up (all offsets 1KB-aligned)
    char* ws = (char*)d_ws;
    size_t o = 0;
    float* lam_re  = (float*)(ws + o); o += NDIM * 4;
    float* lam_im  = (float*)(ws + o); o += NDIM * 4;
    float* lamT_re = (float*)(ws + o); o += NDIM * 4;
    float* lamT_im = (float*)(ws + o); o += NDIM * 4;
    __bf16* Wre  = (__bf16*)(ws + o); o += (size_t)HDIM * NDIM * 2;
    __bf16* Wim  = (__bf16*)(ws + o); o += (size_t)HDIM * NDIM * 2;
    __bf16* CTre = (__bf16*)(ws + o); o += (size_t)HDIM * NDIM * 2;
    __bf16* CTim = (__bf16*)(ws + o); o += (size_t)HDIM * NDIM * 2;
    float* b_re = (float*)(ws + o); o += (size_t)NCHUNK * NDIM * 4;
    float* b_im = (float*)(ws + o); o += (size_t)NCHUNK * NDIM * 4;
    float* p_re = (float*)(ws + o); o += (size_t)NCHUNK * NDIM * 4;
    float* p_im = (float*)(ws + o); o += (size_t)NCHUNK * NDIM * 4;

    k_prep_lambda<<<1, 256, 0, stream>>>(nu_log, theta_log,
                                         lam_re, lam_im, lamT_re, lamT_im);
    k_prep_tiles<<<(HDIM * NDIM) / 256, 256, 0, stream>>>(
        B_re, B_im, gamma_log, C_re, C_im, Wre, Wim, CTre, CTim);
    k_chunk_carry<<<NCHUNK, 256, 0, stream>>>(x, Wre, Wim, lam_re, lam_im,
                                              b_re, b_im);
    k_prefix<<<1, 256, 0, stream>>>(lamT_re, lamT_im, b_re, b_im, p_re, p_im);
    k_output<<<NCHUNK, 256, 0, stream>>>(x, Wre, Wim, CTre, CTim,
                                         lam_re, lam_im, p_re, p_im, Dv, y);
}
